// ResNetGCN_12000138625293
// MI455X (gfx1250) — compile-verified, matrix operand order checked by reading, not run on hardware
//
#include <hip/hip_runtime.h>
#include <hip/hip_bf16.h>

typedef _Float16 half_t;
typedef __attribute__((ext_vector_type(16))) _Float16 v16h;
typedef __attribute__((ext_vector_type(8)))  _Float16 v8h;
typedef __attribute__((ext_vector_type(8)))  float    v8f;

// ---------------------------------------------------------------------------
// Implicit-GEMM conv (plain GEMM when KH=KW=1) on WMMA f16->f32.
//   in  : f16 NHWC [B,Hin,Win,Cin]
//   wt  : f16 [Cout, K], K = KH*KW*Cin ordered (ky,kx,ci)  (i.e. B^T)
//   out : f16 [M, Cout], M = B*Ho*Wo
// Epilogue: v = acc*scale[n] + bias[n] (+ res[m,n]) (+ relu)
//
// Block: 256 thr = 8 waves, block tile 256(M) x 64(N).
// Each wave: 32(M) x 64(N) -> 8 accumulators; K-loop step 64 (K%64==0 for
// this net) -> 16 WMMA per iteration; each B fragment (from a double-buffered
// LDS tile) is reused by two A-row groups, each A fragment by 4 B fragments.
// Tap decode (k/Cin, t/KW) is wave-uniform per 32-k slice (Cin % 32 == 0),
// so it scalarizes; only adds/compares/address-fma stay per-lane.
// ---------------------------------------------------------------------------
__global__ __launch_bounds__(256)
void k_conv_wmma(const half_t* __restrict__ in, const half_t* __restrict__ wt,
                 const float* __restrict__ scale, const float* __restrict__ bias,
                 const half_t* __restrict__ res, half_t* __restrict__ out,
                 int Hin, int Win, int Cin,
                 int Ho, int Wo, int Cout,
                 int KW, int stride, int pad, int relu, int M, int K)
{
    __shared__ half_t Bs[2][64][72];               // double-buffered 64n x 64k tile
    const int tid  = threadIdx.x;
    const int lane = tid & 31;
    const int wid  = tid >> 5;
    const int m0 = blockIdx.x * 256 + wid * 32;
    const int n0 = blockIdx.y * 64;
    const int HoWo = Ho * Wo;
    const int koff = (lane >> 4) * 8;              // ISA A-layout K split per half-wave

    // ---- per-lane A row decode for the two 16-row groups (once) ----
    bool mv[2]; int ayb[2], axb[2]; long pixb[2];
#pragma unroll
    for (int g = 0; g < 2; ++g) {
        const int mrow = m0 + g * 16 + (lane & 15);
        mv[g] = (mrow < M);
        ayb[g] = 0; axb[g] = 0; pixb[g] = 0;
        if (mv[g]) {
            int ab  = mrow / HoWo;
            int r   = mrow - ab * HoWo;
            int aoy = r / Wo;
            int aox = r - aoy * Wo;
            ayb[g] = aoy * stride - pad;
            axb[g] = aox * stride - pad;
            pixb[g] = (long)ab * Hin;
        }
    }

    // ---- cooperative B staging: thread -> (row 0..63, 16-half chunk) ----
    const int srow = tid >> 2;
    const int scol = (tid & 3) * 16;
    const long wrow = (long)(n0 + srow) * K;

    auto stage = [&](int buf, int kb) {
        const half_t* src = wt + wrow + kb + scol;
        *(v8h*)&Bs[buf][srow][scol]     = *(const v8h*)src;
        *(v8h*)&Bs[buf][srow][scol + 8] = *(const v8h*)(src + 8);
    };

    v8f acc[2][4] = {};

    stage(0, 0);
    int cur = 0;
    for (int kb = 0; kb < K; kb += 64) {
        __syncthreads();                            // staging of `cur` complete
        if (kb + 64 < K) {
            stage(cur ^ 1, kb + 64);                // overlap next tile with compute
            __builtin_prefetch(wt + wrow + kb + 128 + scol, 0, 1);
        }

#pragma unroll
        for (int ks = 0; ks < 2; ++ks) {
            const int ksub = kb + ks * 32;
            // wave-uniform tap decode (scalarizes: ksub is uniform)
            const int t  = ksub / Cin;
            const int c0 = ksub - t * Cin;
            const int ky = t / KW;
            const int kx = t - ky * KW;

            // ---- two A fragments (implicit im2col gather, 2x16B each) ----
            v16h afr[2];
#pragma unroll
            for (int g = 0; g < 2; ++g) {
                const int iy = ayb[g] + ky;
                const int ix = axb[g] + kx;
                const bool v = mv[g] && (unsigned)iy < (unsigned)Hin
                                     && (unsigned)ix < (unsigned)Win;
                const half_t* ap = in + ((pixb[g] + iy) * Win + ix) * (long)Cin + c0 + koff;
                v8h a0 = {}, a1 = {};
                if (v) { a0 = *(const v8h*)ap; a1 = *(const v8h*)(ap + 16); }
#pragma unroll
                for (int e = 0; e < 8; ++e) { afr[g][e] = a0[e]; afr[g][8 + e] = a1[e]; }
            }

            // ---- 4 B fragments, each reused by both A groups: 8 WMMAs ----
#pragma unroll
            for (int f = 0; f < 4; ++f) {
                const int r = f * 16 + (lane & 15);
                v8h b0 = *(const v8h*)&Bs[cur][r][ks * 32 + koff];
                v8h b1 = *(const v8h*)&Bs[cur][r][ks * 32 + koff + 16];
                v16h bfr;
#pragma unroll
                for (int e = 0; e < 8; ++e) { bfr[e] = b0[e]; bfr[8 + e] = b1[e]; }
                acc[0][f] = __builtin_amdgcn_wmma_f32_16x16x32_f16(
                                false, afr[0], false, bfr, (short)0, acc[0][f], false, false);
                acc[1][f] = __builtin_amdgcn_wmma_f32_16x16x32_f16(
                                false, afr[1], false, bfr, (short)0, acc[1][f], false, false);
            }
        }
        cur ^= 1;
    }

    // ---- epilogue: BN fold + residual + relu, f16 store ----
#pragma unroll
    for (int g = 0; g < 2; ++g) {
        const int mb = m0 + g * 16 + ((lane >> 4) * 8);
#pragma unroll
        for (int f = 0; f < 4; ++f) {
            const int nn = n0 + f * 16 + (lane & 15);
            const float s = scale ? scale[nn] : 1.0f;
            const float b = bias  ? bias[nn]  : 0.0f;
#pragma unroll
            for (int r = 0; r < 8; ++r) {
                const int m = mb + r;
                if (m < M) {
                    float v = acc[g][f][r] * s + b;
                    if (res)  v += (float)res[(long)m * Cout + nn];
                    if (relu) v = v > 0.f ? v : 0.f;
                    out[(long)m * Cout + nn] = (half_t)v;
                }
            }
        }
    }
}

// ---------------------------------------------------------------------------
// Stem: 7x7 s2 p3 conv (Cin=3) + BN + ReLU.  x NCHW fp32 -> out NHWC f16.
// ---------------------------------------------------------------------------
__global__ void k_stem(const float* __restrict__ x, const float* __restrict__ w,
                       const float* __restrict__ s, const float* __restrict__ b,
                       half_t* __restrict__ out, long total)
{
    long idx = (long)blockIdx.x * blockDim.x + threadIdx.x;
    if (idx >= total) return;
    int co = (int)(idx & 63);
    long p = idx >> 6;
    int ox = (int)(p % 192); p /= 192;
    int oy = (int)(p % 192);
    int bb = (int)(p / 192);
    float acc = 0.f;
    for (int ci = 0; ci < 3; ++ci)
        for (int ky = 0; ky < 7; ++ky) {
            int iy = oy * 2 - 3 + ky;
            if ((unsigned)iy >= 384u) continue;
            for (int kx = 0; kx < 7; ++kx) {
                int ix = ox * 2 - 3 + kx;
                if ((unsigned)ix >= 384u) continue;
                acc += x[(((long)bb * 3 + ci) * 384 + iy) * 384 + ix]
                     * w[((co * 3 + ci) * 7 + ky) * 7 + kx];
            }
        }
    float v = acc * s[co] + b[co];
    out[idx] = (half_t)(v > 0.f ? v : 0.f);
}

// 3x3 s2 p1 maxpool on NHWC f16: 192->96
__global__ void k_maxpool(const half_t* __restrict__ in, half_t* __restrict__ out, long total)
{
    long idx = (long)blockIdx.x * blockDim.x + threadIdx.x;
    if (idx >= total) return;
    int c = (int)(idx & 63);
    long p = idx >> 6;
    int ox = (int)(p % 96); p /= 96;
    int oy = (int)(p % 96);
    int bb = (int)(p / 96);
    float m = -3.4e38f;
    for (int ky = 0; ky < 3; ++ky) {
        int iy = oy * 2 - 1 + ky;
        if ((unsigned)iy >= 192u) continue;
        for (int kx = 0; kx < 3; ++kx) {
            int ix = ox * 2 - 1 + kx;
            if ((unsigned)ix >= 192u) continue;
            float v = (float)in[((((long)bb * 192 + iy) * 192 + ix) << 6) + c];
            m = v > m ? v : m;
        }
    }
    out[idx] = (half_t)m;
}

// OIHW fp32 -> [O][(ky*KW+kx)*Ci + ci] f16 (matches implicit-gemm K order)
__global__ void k_conv_w_prep(const float* __restrict__ w, half_t* __restrict__ out,
                              int O, int Ci, int KH, int KW, long total)
{
    long idx = (long)blockIdx.x * blockDim.x + threadIdx.x;
    if (idx >= total) return;
    int K = Ci * KH * KW;
    int o = (int)(idx / K);
    int k = (int)(idx - (long)o * K);
    int t  = k / Ci;
    int ci = k - t * Ci;
    int ky = t / KW;
    int kx = t - ky * KW;
    out[idx] = (half_t)w[(((long)o * Ci + ci) * KH + ky) * KW + kx];
}

// GCN weight [K,F] fp32 row-major -> [F,K] f16 (B^T for WMMA GEMM)
__global__ void k_gcn_w_prep(const float* __restrict__ w, half_t* __restrict__ out,
                             int K, int F)
{
    long idx = (long)blockIdx.x * blockDim.x + threadIdx.x;
    if (idx >= (long)K * F) return;
    int f = (int)(idx / K);
    int k = (int)(idx - (long)f * K);
    out[idx] = (half_t)w[(long)k * F + f];
}

__global__ void k_fill_f32(float* p, float v, long n)
{
    long i = (long)blockIdx.x * blockDim.x + threadIdx.x;
    if (i < n) p[i] = v;
}

__global__ void k_deg(const int* __restrict__ ei, float* __restrict__ deg, int E, int N)
{
    int e = blockIdx.x * blockDim.x + threadIdx.x;
    if (e >= E + N) return;
    int d = (e < E) ? ei[E + e] : (e - E);   // dst (edges) + self-loops
    atomicAdd(&deg[d], 1.0f);
}

__global__ void k_rsqrt(float* p, int n)
{
    int i = blockIdx.x * blockDim.x + threadIdx.x;
    if (i < n) p[i] = rsqrtf(p[i]);
}

__global__ void k_gcn_bias_init(float* __restrict__ out, const float* __restrict__ bias,
                                long BN, int F)
{
    long idx = (long)blockIdx.x * blockDim.x + threadIdx.x;
    if (idx >= BN * F) return;
    out[idx] = bias[idx % F];
}

// out[b,dst,f] += dinv[src]*dinv[dst] * xw[b,src,f]   (grid: x = E+N, y = B)
__global__ void k_gcn_scatter(const half_t* __restrict__ xw, const int* __restrict__ ei,
                              const float* __restrict__ dinv, float* __restrict__ out,
                              int E, int N, int F)
{
    int e = blockIdx.x;
    int b = blockIdx.y;
    int f = threadIdx.x;
    int s = (e < E) ? ei[e]     : (e - E);
    int d = (e < E) ? ei[E + e] : (e - E);
    float nrm = dinv[s] * dinv[d];
    float v = (float)xw[((long)b * N + s) * F + f] * nrm;
    atomicAdd(&out[((long)b * N + d) * F + f], v);
}

__global__ void k_f32_to_f16(const float* __restrict__ in, half_t* __restrict__ out, long n)
{
    long i = (long)blockIdx.x * blockDim.x + threadIdx.x;
    if (i < n) out[i] = (half_t)in[i];
}

// mean over N nodes -> FC [F]->[C]; one block per batch image
__global__ void k_mean_fc(const float* __restrict__ g, const float* __restrict__ fcw,
                          const float* __restrict__ fcb, float* __restrict__ out,
                          int N, int F, int C)
{
    __shared__ float mean[512];
    int b = blockIdx.x;
    for (int f = threadIdx.x; f < F; f += blockDim.x) {
        float a = 0.f;
        for (int n = 0; n < N; ++n) a += g[((long)b * N + n) * F + f];
        mean[f] = a / (float)N;
    }
    __syncthreads();
    if (threadIdx.x < (unsigned)C) {
        float a = fcb[threadIdx.x];
        for (int f = 0; f < F; ++f) a += mean[f] * fcw[(long)threadIdx.x * F + f];
        out[b * C + threadIdx.x] = a;
    }
}

// ---------------------------------------------------------------------------
// Host orchestration
// ---------------------------------------------------------------------------
static inline dim3 grid1(long n, int t) { return dim3((unsigned)((n + t - 1) / t)); }

struct BlkP { const float *w1,*s1,*b1,*w2,*s2,*b2,*wd,*sd,*bd; };

extern "C" void kernel_launch(void* const* d_in, const int* in_sizes, int n_in,
                              void* d_out, int out_size, void* d_ws, size_t ws_size,
                              hipStream_t stream)
{
    (void)n_in; (void)out_size; (void)ws_size;
    auto f32 = [&](int i) { return (const float*)d_in[i]; };

    // ---- input unpack (setup_inputs insertion order, trunk flattened DFS) ----
    int ii = 0;
    const float* X  = f32(ii++);                   // [32,3,384,384]
    const int*   EI = (const int*)d_in[ii++];      // [2,E]
    const int    E  = in_sizes[1] / 2;
    const float* stem_w = f32(ii++);
    const float* stem_s = f32(ii++);
    const float* stem_b = f32(ii++);
    BlkP bp[8];
    for (int l = 0; l < 4; ++l)
        for (int j = 0; j < 2; ++j) {
            BlkP& p = bp[l * 2 + j];
            p.w1 = f32(ii++); p.s1 = f32(ii++); p.b1 = f32(ii++);
            p.w2 = f32(ii++); p.s2 = f32(ii++); p.b2 = f32(ii++);
            if (l > 0 && j == 0) { p.wd = f32(ii++); p.sd = f32(ii++); p.bd = f32(ii++); }
            else                 { p.wd = nullptr;   p.sd = nullptr;   p.bd = nullptr;   }
        }
    const float* w_c[3]; const float* b_c[3];
    const float* g_w[3]; const float* g_b[3];
    for (int i = 0; i < 3; ++i) {
        w_c[i] = f32(ii++); b_c[i] = f32(ii++);
        g_w[i] = f32(ii++); g_b[i] = f32(ii++);
    }
    const float* fc_w = f32(ii++);
    const float* fc_b = f32(ii++);

    // ---- workspace arena ----
    char* base = (char*)d_ws;
    size_t off = 0;
    auto alloc = [&](size_t bytes) -> char* {
        char* p = base + off;
        off = (off + bytes + 255) & ~(size_t)255;
        return p;
    };

    auto prep_conv = [&](const float* w, int O, int Ci, int KH, int KW) -> half_t* {
        long n = (long)O * Ci * KH * KW;
        half_t* p = (half_t*)alloc((size_t)n * 2);
        k_conv_w_prep<<<grid1(n, 256), 256, 0, stream>>>(w, p, O, Ci, KH, KW, n);
        return p;
    };
    auto conv_gemm = [&](const half_t* in, const half_t* wt, const float* scale,
                         const float* bias, const half_t* res, half_t* out,
                         int B, int Hin, int Win, int Cin, int KH, int KW,
                         int stride, int pad, int Cout, int relu) {
        int Ho = (Hin + 2 * pad - KH) / stride + 1;
        int Wo = (Win + 2 * pad - KW) / stride + 1;
        long M = (long)B * Ho * Wo;
        int K = KH * KW * Cin;                     // always % 64 == 0 in this net
        dim3 g((unsigned)((M + 255) / 256), (unsigned)(Cout / 64));
        k_conv_wmma<<<g, 256, 0, stream>>>(in, wt, scale, bias, res, out,
                                           Hin, Win, Cin, Ho, Wo, Cout,
                                           KW, stride, pad, relu, (int)M, K);
    };

    // ---- stem + maxpool ----
    const long SO = 32L * 192 * 192 * 64;          // stem out elems
    const long P0 = 32L * 96 * 96 * 64;            // layer1 act elems (largest rotating)
    half_t* actStem = (half_t*)alloc((size_t)SO * 2);
    half_t* bufA    = (half_t*)alloc((size_t)P0 * 2);
    half_t* bufB    = (half_t*)alloc((size_t)P0 * 2);
    half_t* bufY    = (half_t*)alloc((size_t)P0 * 2);
    half_t* bufSC   = (half_t*)alloc((size_t)(32L * 48 * 48 * 128) * 2);

    k_stem<<<grid1(SO, 256), 256, 0, stream>>>(X, stem_w, stem_s, stem_b, actStem, SO);
    k_maxpool<<<grid1(P0, 256), 256, 0, stream>>>(actStem, bufA, P0);

    // ---- residual layers (implicit-GEMM WMMA convs) ----
    const int cfg_c[4] = {64, 128, 256, 512};
    const int cfg_s[4] = {1, 2, 2, 2};
    half_t* x = bufA;
    half_t* other = bufB;
    int H = 96, Cin = 64;
    for (int l = 0; l < 4; ++l)
        for (int j = 0; j < 2; ++j) {
            const BlkP& bk = bp[l * 2 + j];
            const int s    = (j == 0) ? cfg_s[l] : 1;
            const int Cout = cfg_c[l];
            const int Hout = H / s;
            half_t* w1 = prep_conv(bk.w1, Cout, Cin, 3, 3);
            half_t* w2 = prep_conv(bk.w2, Cout, Cout, 3, 3);
            // y1 = relu(bn1(conv3x3(x, s)))
            conv_gemm(x, w1, bk.s1, bk.b1, nullptr, bufY,
                      32, H, H, Cin, 3, 3, s, 1, Cout, 1);
            // shortcut
            const half_t* sc = x;
            if (bk.wd) {
                half_t* wd = prep_conv(bk.wd, Cout, Cin, 1, 1);
                conv_gemm(x, wd, bk.sd, bk.bd, nullptr, bufSC,
                          32, H, H, Cin, 1, 1, s, 0, Cout, 0);
                sc = bufSC;
            }
            // out = relu(bn2(conv3x3(y1)) + sc)
            conv_gemm(bufY, w2, bk.s2, bk.b2, sc, other,
                      32, Hout, Hout, Cout, 3, 3, 1, 1, Cout, 1);
            half_t* t = x; x = other; other = t;
            H = Hout; Cin = Cout;
        }
    // x: f16 NHWC [32,12,12,512]

    // ---- GCN head ----
    const int N = 144;
    float* deg = (float*)alloc(N * 4);
    k_fill_f32<<<grid1(N, 64), 64, 0, stream>>>(deg, 0.f, N);
    k_deg<<<grid1(E + N, 256), 256, 0, stream>>>(EI, deg, E, N);
    k_rsqrt<<<grid1(N, 64), 64, 0, stream>>>(deg, N);

    const long GNF = 32L * N * 512;
    half_t* t1  = (half_t*)alloc((size_t)GNF * 2);
    half_t* t2  = (half_t*)alloc((size_t)GNF * 2);
    half_t* gxn = (half_t*)alloc((size_t)GNF * 2);
    float*  g32 = (float*)alloc((size_t)GNF * 4);

    const int Fs[3] = {128, 256, 512};
    half_t* gx = x;
    int Cg = 512;
    for (int i = 0; i < 3; ++i) {
        const int F = Fs[i];
        half_t* wc = prep_conv(w_c[i], F, Cg, 1, 1);
        // conv1x1 + bias
        conv_gemm(gx, wc, nullptr, b_c[i], nullptr, t1,
                  32, 12, 12, Cg, 1, 1, 1, 0, F, 0);
        // xw = t1 @ W
        half_t* wg = (half_t*)alloc((size_t)F * F * 2);
        k_gcn_w_prep<<<grid1((long)F * F, 256), 256, 0, stream>>>(g_w[i], wg, F, F);
        conv_gemm(t1, wg, nullptr, nullptr, nullptr, t2,
                  32, 12, 12, F, 1, 1, 1, 0, F, 0);
        // out = bias + scatter(norm * gather(xw))
        k_gcn_bias_init<<<grid1(32L * N * F, 256), 256, 0, stream>>>(g32, g_b[i], 32L * N, F);
        k_gcn_scatter<<<dim3(E + N, 32), F, 0, stream>>>(t2, EI, deg, g32, E, N, F);
        if (i < 2) {
            k_f32_to_f16<<<grid1(32L * N * F, 256), 256, 0, stream>>>(g32, gxn, 32L * N * F);
            gx = gxn; Cg = F;
        }
    }

    // ---- global mean pool + FC -> d_out [32,3] fp32 ----
    k_mean_fc<<<32, 256, 0, stream>>>(g32, fc_w, fc_b, (float*)d_out, N, 512, 3);
}
